// RNN_8220567404960
// MI455X (gfx1250) — compile-verified, matrix operand order checked by reading, not run on hardware
//
#include <hip/hip_runtime.h>

// ---------------------------------------------------------------------------
// LSTM char-RNN on gfx1250 (MI455X).  B=4096, T=40, V=128, H=256.
// z = W_lstm[char,:] + h @ W_h + b  (one-hot x => row gather), gates i,j,f,o.
//
// Round 3 = round 2 +, for B staging, GLOBAL_LOAD_ASYNC_TO_LDS_B128
// (ASYNCcnt-tracked DMA into LDS, no VGPR round-trip) with explicit
// s_wait_asynccnt 0 before the consumption barrier.
//
// Mapping:
//  * WG = 256 threads (8 waves), owns 64 batch rows (4 M-tiles). Grid = 64.
//  * wave w: M-tile (w>>1), hidden half (w&1) -> 8 hidden chunks of 16.
//  * Recurrent GEMM: v_wmma_f32_16x16x32_bf16, A (h tile) resident in VGPRs,
//    B (packed bf16 W_h) async-staged global->LDS, double buffered: each
//    64KB tile-pair loaded once per WG per step, consumed by all 8 waves.
//  * c state fp32 in LDS, h state bf16 in LDS (LDS provides D->A transpose).
//  * Gates processed sequentially (one live accumulator) -> no VGPR spills.
//  * LDS: h 32KB + c 64KB + B 2x64KB = 224KB of 320KB/WGP.
// ---------------------------------------------------------------------------

#define RNN_B 4096
#define RNN_T 40
#define RNN_V 128
#define RNN_H 256
#define RNN_4H 1024

typedef __attribute__((ext_vector_type(16))) __bf16 v16bf;
typedef __attribute__((ext_vector_type(8)))  float  v8f;

union FragU {
    uint4 q[2];
    v16bf v;
};

__device__ __forceinline__ unsigned bf_bits(float f) {
    union { float f; unsigned u; } a;
    a.f = f;
    unsigned r = a.u + 0x7FFFu + ((a.u >> 16) & 1u);   // fp32 -> bf16 RNE
    return r >> 16;
}

__device__ __forceinline__ float fsig(float x) {
    return 1.0f / (1.0f + __expf(-x));
}
__device__ __forceinline__ float ftanh(float x) {
    return 1.0f - 2.0f / (__expf(2.0f * x) + 1.0f);    // saturates at +-1
}

// Async DMA: 16B global -> LDS, tracked by ASYNCcnt (no VGPR data round-trip).
// GV addressing: vdst = LDS byte offset, v[addr] = 64-bit global address.
__device__ __forceinline__ void async_b128_to_lds(unsigned lds_off, const void* gptr) {
    asm volatile("global_load_async_to_lds_b128 %0, %1, off"
                 :: "v"(lds_off), "v"((unsigned long long)(uintptr_t)gptr)
                 : "memory");
}
__device__ __forceinline__ void wait_async0() {
    asm volatile("s_wait_asynccnt 0x0" ::: "memory");
}

// ---------------------------------------------------------------------------
// Pack W_h (= W_lstm rows 128..383, [256 x 1024] row-major fp32) into bf16
// WMMA B fragments, ordered exactly in per-step consumption order:
//   pair i (0..7) = the two hidden chunks {i, 8+i}; within a pair:
//   tile = hh*32 + gi*8 + kb  (hh = hidden half, gi = gate, kb = K block).
// dword index = ((pair*64 + tile)*32 + lane)*8 + d
// Per-lane dword d holds K = kb*32 + (lane>=16?8:0) + (d<4 ? 2d : 16+2(d-4)),
// packed with K+1, N = (gi*16 + hh*8 + pair)*16 + (lane&15).
// Lane's 8 dwords are contiguous -> fragment fetch = 2 x ds_load_b128.
// ---------------------------------------------------------------------------
__global__ void pack_wh_kernel(const float* __restrict__ Wl, unsigned* __restrict__ Wp) {
    int tid = blockIdx.x * blockDim.x + threadIdx.x;
    if (tid >= 512 * 256) return;                 // 512 tiles * 256 dwords
    int d    = tid & 7;
    int lane = (tid >> 3) & 31;
    int tIdx = tid >> 8;                          // 0..511
    int tile = tIdx & 63;
    int pair = tIdx >> 6;                         // 0..7
    int kb   = tile & 7;
    int gi   = (tile >> 3) & 3;
    int hh   = tile >> 5;
    int hc   = hh * 8 + pair;                     // hidden chunk 0..15
    int nt   = gi * 16 + hc;
    int n    = nt * 16 + (lane & 15);
    int k0   = kb * 32 + ((lane >> 4) * 8) + ((d < 4) ? 2 * d : 16 + 2 * (d - 4));
    float f0 = Wl[(RNN_V + k0    ) * RNN_4H + n];
    float f1 = Wl[(RNN_V + k0 + 1) * RNN_4H + n];
    Wp[tid] = bf_bits(f0) | (bf_bits(f1) << 16);
}

// W_dense [256 x 128] row-major fp32, nt-major packing (consumed per-wave).
__global__ void pack_wd_kernel(const float* __restrict__ Wd, unsigned* __restrict__ Wp) {
    int tid = blockIdx.x * blockDim.x + threadIdx.x;
    if (tid >= 64 * 256) return;                  // 64 tiles * 256 dwords
    int d    = tid & 7;
    int lane = (tid >> 3) & 31;
    int tile = tid >> 8;
    int kb   = tile & 7;
    int nt   = tile >> 3;
    int n    = nt * 16 + (lane & 15);
    int k0   = kb * 32 + ((lane >> 4) * 8) + ((d < 4) ? 2 * d : 16 + 2 * (d - 4));
    float f0 = Wd[(k0    ) * RNN_V + n];
    float f1 = Wd[(k0 + 1) * RNN_V + n];
    Wp[tid] = bf_bits(f0) | (bf_bits(f1) << 16);
}

// ---------------------------------------------------------------------------
__global__ __launch_bounds__(256, 1)
void lstm_rnn_kernel(const int*   __restrict__ inp,
                     const float* __restrict__ Wl,
                     const float* __restrict__ bl,
                     const float* __restrict__ bd,
                     const uint4* __restrict__ WpH,
                     const uint4* __restrict__ WpD,
                     float* __restrict__ out) {
    __shared__ unsigned short hbuf[4][16][RNN_H];  // bf16 h per M-tile, 32KB
    __shared__ float          cbuf[8][16][128];    // fp32 c per wave,    64KB
    __shared__ uint4          bstage[2][4096];     // B double buffer,   128KB

    const int tid  = threadIdx.x;
    const int lane = tid & 31;
    const int wave = tid >> 5;
    const int mt   = wave >> 1;        // M-tile 0..3
    const int hh   = wave & 1;         // hidden half
    const int s    = lane >> 4;        // half-wave select
    const int mlo  = lane & 15;
    const int b0   = blockIdx.x * 64 + mt * 16;

    for (int idx = tid; idx < 4 * 16 * RNN_H; idx += 256) ((unsigned short*)hbuf)[idx] = 0;
    for (int idx = tid; idx < 8 * 16 * 128; idx += 256) ((float*)cbuf)[idx] = 0.0f;

    // stage pair 0 (async DMA to LDS)
    {
        const unsigned dst0 = (unsigned)(uintptr_t)(&bstage[0][tid]);
        const uint4* src0 = WpH + tid;
#pragma unroll
        for (int q = 0; q < 16; ++q)
            async_b128_to_lds(dst0 + (unsigned)(q * 4096), src0 + 256 * q);
    }
    wait_async0();
    __syncthreads();

    FragU a[8];

    for (int t = 0; t < RNN_T; ++t) {
        int cm[8];
#pragma unroll
        for (int r = 0; r < 8; ++r)
            cm[r] = inp[(b0 + r + 8 * s) * RNN_T + t];

        // A fragments: h tile of this M-tile (read before anyone rewrites it)
#pragma unroll
        for (int kb = 0; kb < 8; ++kb) {
            a[kb].q[0] = *(const uint4*)&hbuf[mt][mlo][kb * 32 + 8 * s];
            a[kb].q[1] = *(const uint4*)&hbuf[mt][mlo][kb * 32 + 16 + 8 * s];
        }
        __syncthreads();

#pragma unroll 1
        for (int i = 0; i < 8; ++i) {
            const int p = i & 1;
            // async-prefetch next tile-pair into the other buffer
            {
                const int nx = (i + 1) & 7;
                const unsigned dst0 = (unsigned)(uintptr_t)(&bstage[p ^ 1][tid]);
                const uint4* src0 = WpH + nx * 4096 + tid;
#pragma unroll
                for (int q = 0; q < 16; ++q)
                    async_b128_to_lds(dst0 + (unsigned)(q * 4096), src0 + 256 * q);
            }
            const int hc = hh * 8 + i;

            // one gate's 16x16 z-tile: 8 WMMAs from LDS-staged B + x gather
            auto gate_mm = [&](int gi) -> v8f {
                v8f acc;
#pragma unroll
                for (int r = 0; r < 8; ++r) acc[r] = 0.0f;
#pragma unroll
                for (int kb = 0; kb < 8; ++kb) {
                    FragU bf;
                    const uint4* bp = &bstage[p][(hh * 32 + gi * 8 + kb) * 64 + lane * 2];
                    bf.q[0] = bp[0];
                    bf.q[1] = bp[1];
                    acc = __builtin_amdgcn_wmma_f32_16x16x32_bf16(
                        false, a[kb].v, false, bf.v, (short)0, acc, false, false);
                }
                const int n = gi * 256 + hc * 16 + mlo;
                const float bias = bl[n];
#pragma unroll
                for (int r = 0; r < 8; ++r)
                    acc[r] += Wl[cm[r] * RNN_4H + n] + bias;
                return acc;
            };

            // sequential gates: at most one acc + one temp live at a time
            v8f zi = gate_mm(0);
            v8f zj = gate_mm(1);
            v8f tmp;
#pragma unroll
            for (int r = 0; r < 8; ++r) tmp[r] = fsig(zi[r]) * ftanh(zj[r]);

            v8f zf = gate_mm(2);
            const int lc = i * 16 + mlo;           // wave-local c column
            v8f cn;
#pragma unroll
            for (int r = 0; r < 8; ++r) {
                float cold = cbuf[wave][r + 8 * s][lc];
                float c2 = fsig(zf[r] + 1.0f) * cold + tmp[r];
                cn[r] = c2;
                cbuf[wave][r + 8 * s][lc] = c2;
            }

            v8f zo = gate_mm(3);
            const int n = hc * 16 + mlo;
#pragma unroll
            for (int r = 0; r < 8; ++r)
                hbuf[mt][r + 8 * s][n] =
                    (unsigned short)bf_bits(fsig(zo[r]) * ftanh(cn[r]));

            wait_async0();     // prefetch of pair i+1 fully in LDS
            __syncthreads();   // + pair i consumed by all waves
        }
    }

    // ---- final dense: out = h @ W_dense + b_dense ----
#pragma unroll
    for (int kb = 0; kb < 8; ++kb) {
        a[kb].q[0] = *(const uint4*)&hbuf[mt][mlo][kb * 32 + 8 * s];
        a[kb].q[1] = *(const uint4*)&hbuf[mt][mlo][kb * 32 + 16 + 8 * s];
    }
#pragma unroll 1
    for (int j = 0; j < 4; ++j) {
        const int nt = hh * 4 + j;                 // 0..7 (V = 128)
        v8f acc;
#pragma unroll
        for (int r = 0; r < 8; ++r) acc[r] = 0.0f;
#pragma unroll
        for (int kb = 0; kb < 8; ++kb) {
            FragU bf;
            const uint4* bp = WpD + ((nt * 8 + kb) * 32 + lane) * 2;
            bf.q[0] = bp[0];
            bf.q[1] = bp[1];
            acc = __builtin_amdgcn_wmma_f32_16x16x32_bf16(
                false, a[kb].v, false, bf.v, (short)0, acc, false, false);
        }
        const int n = nt * 16 + mlo;
        const float bias = bd[n];
#pragma unroll
        for (int r = 0; r < 8; ++r)
            out[(b0 + r + 8 * s) * RNN_V + n] = acc[r] + bias;
    }
}

extern "C" void kernel_launch(void* const* d_in, const int* in_sizes, int n_in,
                              void* d_out, int out_size, void* d_ws, size_t ws_size,
                              hipStream_t stream) {
    const int*   inp = (const int*)  d_in[0];   // [4096, 40] int32
    const float* Wl  = (const float*)d_in[1];   // [384, 1024]
    const float* bl  = (const float*)d_in[2];   // [1024]
    const float* Wd  = (const float*)d_in[3];   // [256, 128]
    const float* bd  = (const float*)d_in[4];   // [128]
    float* out = (float*)d_out;                 // [4096, 128] fp32

    unsigned* WpH = (unsigned*)d_ws;            // 131072 dwords = 512 KB
    unsigned* WpD = WpH + 512 * 256;            //  16384 dwords =  64 KB

    pack_wh_kernel<<<(512 * 256 + 255) / 256, 256, 0, stream>>>(Wl, WpH);
    pack_wd_kernel<<<(64 * 256 + 255) / 256, 256, 0, stream>>>(Wd, WpD);

    lstm_rnn_kernel<<<RNN_B / 64, 256, 0, stream>>>(
        inp, Wl, bl, bd, (const uint4*)WpH, (const uint4*)WpD, out);
}
